// SparseOrthogonalBasisSelfAttention_17712445129299
// MI455X (gfx1250) — matrix operand order repaired and checked by reading, not compiled, not run on hardware
//
#include <hip/hip_runtime.h>
#include <hip/hip_bf16.h>

#define B_    2
#define L_    2048
#define HID_  1024
#define H_    16
#define D_    64
#define NB_   6
#define SMB_  64    // staged m-block (columns) per LDS buffer

typedef __attribute__((ext_vector_type(16))) __bf16 v16bf;
typedef __attribute__((ext_vector_type(8)))  float  v8f;
typedef __attribute__((ext_vector_type(4)))  float  f32x4;

union FragU { v16bf v; f32x4 q[2]; unsigned u[8]; };

static __device__ inline v8f wmma_bf16(v16bf a, v16bf b, v8f c) {
  return __builtin_amdgcn_wmma_f32_16x16x32_bf16(false, a, false, b, (short)0, c, false, false);
}
static __device__ inline v16bf ldfrag(const __bf16* p0, const __bf16* p1) {
  FragU f;
  f.q[0] = *(const f32x4*)p0;
  f.q[1] = *(const f32x4*)p1;
  return f.v;
}
static __device__ inline float swz16f(float x) {  // SWAPX16: lane <-> lane^16
  return __int_as_float(__builtin_amdgcn_ds_swizzle(__float_as_int(x), 0x401F));
}
static __device__ inline unsigned swz16u(unsigned x) {
  return (unsigned)__builtin_amdgcn_ds_swizzle((int)x, 0x401F);
}
static __device__ inline unsigned pkbf(float a, float b) {
  union { __bf16 h[2]; unsigned u; } r;
  r.h[0] = (__bf16)a; r.h[1] = (__bf16)b;
  return r.u;
}
// CDNA5 async global->LDS copy (16B per lane), tracked by ASYNCcnt
static __device__ inline void async_ld_b128(unsigned lds_off, const __bf16* gp) {
  unsigned long long ga = (unsigned long long)(size_t)gp;
  asm volatile("global_load_async_to_lds_b128 %0, %1, off" :: "v"(lds_off), "v"(ga) : "memory");
}
static __device__ inline void wait_async0() {
  asm volatile("s_wait_asynccnt 0x0" ::: "memory");
}

// ---------------- conversion / small precompute kernels ----------------

__global__ void k_cvt(const float* __restrict__ x, __bf16* __restrict__ y, int n) {
  int i = blockIdx.x * blockDim.x + threadIdx.x;
  if (i < n) y[i] = (__bf16)x[i];
}

__global__ void k_transpose(const float* __restrict__ w, __bf16* __restrict__ wt) {
  int i = blockIdx.x * blockDim.x + threadIdx.x;   // 0..1M-1
  int n = i >> 10, k = i & 1023;
  wt[(size_t)n * 1024 + k] = (__bf16)w[(size_t)k * 1024 + n];
}

// Mt[h][d2][d1] = (1/sqrt(512)) * sum_r Qw[h,d1,r] * Kw[h,d2,r]
__global__ void k_mproj(const float* __restrict__ qw, const float* __restrict__ kw,
                        __bf16* __restrict__ mt) {
  int i = blockIdx.x * blockDim.x + threadIdx.x;   // H*64*64
  int h = i >> 12, d2 = (i >> 6) & 63, d1 = i & 63;
  const float* qr = qw + ((size_t)h * 64 + d1) * 512;
  const float* kr = kw + ((size_t)h * 64 + d2) * 512;
  float s = 0.f;
  for (int r = 0; r < 512; ++r) s += qr[r] * kr[r];
  mt[i] = (__bf16)(s * 0.04419417382415922f);      // 1/sqrt(512)
}

// bias table[h][d], d = |i-j| in 0..63
__global__ void k_bias(const float* __restrict__ coeff, float* __restrict__ table) {
  int i = blockIdx.x * blockDim.x + threadIdx.x;
  if (i >= H_ * 64) return;
  int h = i >> 6, d = i & 63;
  const float PI = 3.14159265358979323846f;
  float s = 0.f;
  for (int k = 0; k < NB_; ++k) {
    float sc = (k == 0) ? __fsqrt_rn(1.0f / L_) : __fsqrt_rn(2.0f / L_);
    s += coeff[h * NB_ + k] * sc * __cosf(PI * ((float)d + 0.5f) * (float)k / (float)L_);
  }
  table[i] = s;
}

// ---- QKV GEMM: X[4096,1024] x Wt -> Q/K [B,H,L,64], V [B,H,64,L] (bf16) ----
// 32x64 output per wave; ping-pong fragment register sets (no rotation copies,
// even-set prefetch guarded in loop): loads for one set issue before the other's WMMAs.

__global__ __launch_bounds__(256) void k_gemm(
    const __bf16* __restrict__ x,
    const __bf16* __restrict__ wt0, const __bf16* __restrict__ wt1, const __bf16* __restrict__ wt2,
    const float* __restrict__ b0, const float* __restrict__ b1, const float* __restrict__ b2,
    __bf16* __restrict__ outq, __bf16* __restrict__ outk, __bf16* __restrict__ outv) {
  int lane = threadIdx.x & 31, wave = threadIdx.x >> 5;
  int rt  = blockIdx.x * 8 + wave;   // 32-row tile 0..127
  int n64 = blockIdx.y;              // 0..15
  int mat = blockIdx.z;              // 0..2
  const __bf16* wt = (mat == 0) ? wt0 : (mat == 1) ? wt1 : wt2;
  const float* bias = (mat == 0) ? b0 : (mat == 1) ? b1 : b2;
  int n16 = lane & 15;
  int hi8 = (lane >> 4) * 8, hi16 = hi8 * 2;

  const __bf16* xr0 = x + (size_t)(rt * 32 + n16) * 1024 + hi8;
  const __bf16* xr1 = xr0 + (size_t)16 * 1024;
  const __bf16* wr  = wt + (size_t)(n64 * 64 + n16) * 1024 + hi8;

  v8f zz = {0.f,0.f,0.f,0.f,0.f,0.f,0.f,0.f};
  v8f c[2][4] = {{zz, zz, zz, zz}, {zz, zz, zz, zz}};

  v16bf aA[2], bA[4], aB[2], bB[4];
#define LOADSET(K, AA, BB)                                              \
  do {                                                                  \
    AA[0] = ldfrag(xr0 + (K), xr0 + (K) + 16);                          \
    AA[1] = ldfrag(xr1 + (K), xr1 + (K) + 16);                          \
    for (int nt = 0; nt < 4; ++nt) {                                    \
      const __bf16* bp = wr + (size_t)nt * 16 * 1024 + (K);             \
      BB[nt] = ldfrag(bp, bp + 16);                                     \
    }                                                                   \
  } while (0)

  LOADSET(0, aA, bA);
  for (int k0 = 0; k0 < 1024; k0 += 64) {
    LOADSET(k0 + 32, aB, bB);                       // prefetch odd set
    for (int at = 0; at < 2; ++at)
      for (int nt = 0; nt < 4; ++nt)
        c[at][nt] = wmma_bf16(aA[at], bA[nt], c[at][nt]);
    if (k0 + 64 < 1024) LOADSET(k0 + 64, aA, bA);   // prefetch even set
    for (int at = 0; at < 2; ++at)
      for (int nt = 0; nt < 4; ++nt)
        c[at][nt] = wmma_bf16(aB[at], bB[nt], c[at][nt]);
  }
#undef LOADSET

  for (int at = 0; at < 2; ++at)
    for (int nt = 0; nt < 4; ++nt) {
      int col = n64 * 64 + nt * 16 + n16;
      int h = col >> 6, d = col & 63;
      float bv = bias[col];
      for (int v = 0; v < 8; ++v) {
        int row = rt * 32 + at * 16 + v + hi8;
        int b = row >> 11, l = row & 2047;
        float val = c[at][nt][v] + bv;
        if (mat == 2)
          outv[(((size_t)b * H_ + h) * 64 + d) * L_ + l] = (__bf16)val;     // V transposed
        else {
          __bf16* o = (mat == 0) ? outq : outk;
          o[(((size_t)b * H_ + h) * L_ + l) * 64 + d] = (__bf16)val;
        }
      }
    }
}

// ---------------- fold: Qp[b,h,l,d2] = sum_d1 Qtmp[b,h,l,d1] * M[d1,d2] ----------------

__global__ __launch_bounds__(128) void k_fold(const __bf16* __restrict__ qtmp,
                                              const __bf16* __restrict__ mt,
                                              __bf16* __restrict__ qp) {
  int lane = threadIdx.x & 31, wave = threadIdx.x >> 5;
  int lt = blockIdx.x * 4 + wave;    // 0..127
  int h = blockIdx.y, b = blockIdx.z;
  int n16 = lane & 15;
  int hi8 = (lane >> 4) * 8, hi16 = hi8 * 2;
  const __bf16* qh = qtmp + (((size_t)b * H_ + h) * L_) * 64;
  const __bf16* mh = mt + (size_t)h * 64 * 64;
  __bf16* oh = qp + (((size_t)b * H_ + h) * L_) * 64;

  v16bf a[2];
  for (int kh = 0; kh < 2; ++kh) {
    const __bf16* ap = qh + (size_t)(lt * 16 + n16) * 64 + kh * 32 + hi8;
    a[kh] = ldfrag(ap, ap + 16);
  }
  v8f zz = {0.f,0.f,0.f,0.f,0.f,0.f,0.f,0.f};
  for (int nt = 0; nt < 4; ++nt) {
    v8f c = zz;
    for (int kh = 0; kh < 2; ++kh) {
      const __bf16* bp = mh + (size_t)(nt * 16 + n16) * 64 + kh * 32 + hi8;
      c = wmma_bf16(a[kh], ldfrag(bp, bp + 16), c);
    }
    for (int v = 0; v < 8; ++v)
      oh[(size_t)(lt * 16 + v + hi8) * 64 + nt * 16 + n16] = (__bf16)c[v];
  }
}

// ---- fused flash attention; K/V^T staged via async global->LDS, double buffered ----

__global__ __launch_bounds__(256) void k_attn(const __bf16* __restrict__ qp,
                                              const __bf16* __restrict__ kk,
                                              const __bf16* __restrict__ vt,
                                              const float* __restrict__ table,
                                              float* __restrict__ out) {
  __shared__ __align__(16) __bf16 sK[2][SMB_ * 64];  // [m_local][d]
  __shared__ __align__(16) __bf16 sV[2][64 * SMB_];  // [d][m_local]

  int tid  = threadIdx.x;
  int lane = tid & 31, wave = tid >> 5;
  int qt = blockIdx.x * 8 + wave;    // 0..127
  int h = blockIdx.y, b = blockIdx.z;
  int n16 = lane & 15;
  int hi8 = (lane >> 4) * 8, hi16 = hi8 * 2;
  size_t hb = (size_t)b * H_ + h;
  const __bf16* qh = qp + hb * L_ * 64;
  const __bf16* kh = kk + hb * L_ * 64;
  const __bf16* vh = vt + hb * 64 * L_;
  const float* tb = table + h * 64;

  // q' as B-operand fragments (K = d), loop-invariant, from global
  v16bf qb[2];
  for (int khf = 0; khf < 2; ++khf) {
    const __bf16* p = qh + (size_t)(qt * 16 + n16) * 64 + khf * 32 + hi8;
    qb[khf] = ldfrag(p, p + 16);
  }

  int i = qt * 16 + n16;                     // this lane's query row
  int lo = i - 32; if (lo < 0) lo = 0; if (lo > L_ - 64) lo = L_ - 64;
  int lo_min = qt * 16 - 32;      if (lo_min < 0) lo_min = 0; if (lo_min > L_ - 64) lo_min = L_ - 64;
  int lo_max = qt * 16 + 15 - 32; if (lo_max < 0) lo_max = 0; if (lo_max > L_ - 64) lo_max = L_ - 64;
  int jhi = lo_max + 63;

  // cooperative async stage of one 64-col K tile + V^T tile (8KB each)
  auto stage = [&](int it, int buf) {
    int mb = it * SMB_;
#pragma unroll
    for (int t = 0; t < 2; ++t) {
      int cch = tid + t * 256;
      int row = cch >> 3, col = (cch & 7) * 8;      // 16B chunks, 8 per row
      async_ld_b128((unsigned)(size_t)&sK[buf][row * 64 + col],
                    kh + (size_t)(mb + row) * 64 + col);
      async_ld_b128((unsigned)(size_t)&sV[buf][row * 64 + col],
                    vh + (size_t)row * L_ + mb + col);
    }
  };

  v8f zz = {0.f,0.f,0.f,0.f,0.f,0.f,0.f,0.f};
  v8f acc[4] = {zz, zz, zz, zz};
  float rmax = -3.0e38f, rsum = 0.f;

  stage(0, 0);
  const int NIT = L_ / SMB_;                 // 32
  for (int it = 0; it < NIT; ++it) {
    int buf = it & 1;
    wait_async0();                           // this wave's stage(it) chunks done
    __syncthreads();                         // whole tile visible; prev buf free
    if (it + 1 < NIT) stage(it + 1, buf ^ 1);

    for (int half = 0; half < 2; ++half) {
      int mb = it * SMB_ + half * 32;        // global m base
      int ml = half * 32;                    // LDS-local m base
      // S^T tiles [32m x 16q], two 16x16 tiles, K = d = 64, A = k from LDS
      v8f s[2];
      for (int t = 0; t < 2; ++t) {
        v8f sc = zz;
        int mloc = ml + t * 16 + n16;
        for (int khf = 0; khf < 2; ++khf) {
          const __bf16* ap = &sK[buf][mloc * 64 + khf * 32 + hi8];
          sc = wmma_bf16(ldfrag(ap, ap + 16), qb[khf], sc);
        }
        s[t] = sc;
      }
      // banded distance bias (scale 1/sqrt(R) already folded into M)
      if (mb + 31 >= lo_min && mb <= jhi) {
        for (int t = 0; t < 2; ++t)
          for (int v = 0; v < 8; ++v) {
            int j = mb + t * 16 + v + hi8;
            if (j >= lo && j < lo + 64) {
              int dd = j - i; if (dd < 0) dd = -dd;
              s[t][v] += tb[dd];
            }
          }
      }
      // online softmax; query row lives in lane n and lane n+16
      float bm = s[0][0];
      for (int t = 0; t < 2; ++t)
        for (int v = 0; v < 8; ++v) bm = fmaxf(bm, s[t][v]);
      bm = fmaxf(bm, swz16f(bm));
      float nm = fmaxf(rmax, bm);
      float alpha = __expf(rmax - nm);
      rmax = nm;
      rsum *= alpha;
      for (int t = 0; t < 4; ++t) acc[t] = acc[t] * alpha;

      float pe[2][8];
      float psum = 0.f;
      for (int t = 0; t < 2; ++t)
        for (int v = 0; v < 8; ++v) { pe[t][v] = __expf(s[t][v] - nm); psum += pe[t][v]; }
      rsum += psum;

      // pack probs into P^T B-operand layout (pack + SWAPX16 + half-select)
      FragU pb;
      for (int bb = 0; bb < 4; ++bb) {
        unsigned p0 = pkbf(pe[0][2 * bb], pe[0][2 * bb + 1]);
        unsigned p1 = pkbf(pe[1][2 * bb], pe[1][2 * bb + 1]);
        unsigned x1 = swz16u(p1);
        unsigned x0 = swz16u(p0);
        pb.u[bb]     = (lane < 16) ? p0 : x1;
        pb.u[bb + 4] = (lane < 16) ? x0 : p1;
      }
      // out^T[d, q] += V^T[16d x 32m] * P^T[32m x 16q], A = V^T from LDS
      for (int db = 0; db < 4; ++db) {
        const __bf16* ap = &sV[buf][(db * 16 + n16) * 64 + ml + hi16];
        acc[db] = wmma_bf16(ldfrag(ap, ap + 8), pb.v, acc[db]);
      }
    }
  }

  float tot = rsum + swz16f(rsum);
  float rinv = 1.0f / tot;
  for (int db = 0; db < 4; ++db)
    for (int v = 0; v < 8; ++v) {
      int d = db * 16 + v + hi8;
      out[((size_t)b * L_ + i) * HID_ + h * 64 + d] = acc[db][v] * rinv;
    }
}

// ---------------- host launcher ----------------

extern "C" void kernel_launch(void* const* d_in, const int* in_sizes, int n_in,
                              void* d_out, int out_size, void* d_ws, size_t ws_size,
                              hipStream_t stream) {
  (void)in_sizes; (void)n_in; (void)out_size; (void)ws_size;
  const float* hs  = (const float*)d_in[0];
  const float* Wq  = (const float*)d_in[1];
  const float* bq  = (const float*)d_in[2];
  const float* Wk  = (const float*)d_in[3];
  const float* bk  = (const float*)d_in[4];
  const float* Wv  = (const float*)d_in[5];
  const float* bv  = (const float*)d_in[6];
  const float* cf  = (const float*)d_in[7];
  const float* qpw = (const float*)d_in[8];
  const float* kpw = (const float*)d_in[9];
  float* out = (float*)d_out;

  char* ws = (char*)d_ws;
  __bf16* Xbf  = (__bf16*)(ws + 0);                       //  8 MB  X bf16
  __bf16* WT0  = (__bf16*)(ws + 8388608);                 //  2 MB  Wq^T
  __bf16* WT1  = (__bf16*)(ws + 10485760);                //  2 MB  Wk^T
  __bf16* WT2  = (__bf16*)(ws + 12582912);                //  2 MB  Wv^T
  __bf16* QT   = (__bf16*)(ws + 14680064);                //  8 MB  Q  [B,H,L,64]
  __bf16* KB   = (__bf16*)(ws + 23068672);                //  8 MB  K  [B,H,L,64]
  __bf16* VT   = (__bf16*)(ws + 31457280);                //  8 MB  V^T[B,H,64,L]
  __bf16* QP   = (__bf16*)(ws + 39845888);                //  8 MB  Q' [B,H,L,64]
  __bf16* MT   = (__bf16*)(ws + 48234496);                // 128 KB M^T bf16
  float*  TBL  = (float*)(ws + 48365568);                 //  4 KB  bias table

  k_cvt<<<(B_ * L_ * HID_) / 256, 256, 0, stream>>>(hs, Xbf, B_ * L_ * HID_);
  k_transpose<<<4096, 256, 0, stream>>>(Wq, WT0);
  k_transpose<<<4096, 256, 0, stream>>>(Wk, WT1);
  k_transpose<<<4096, 256, 0, stream>>>(Wv, WT2);
  k_mproj<<<(H_ * 64 * 64) / 256, 256, 0, stream>>>(qpw, kpw, MT);
  k_bias<<<4, 256, 0, stream>>>(cf, TBL);
  k_gemm<<<dim3(16, 16, 3), 256, 0, stream>>>(Xbf, WT0, WT1, WT2, bq, bk, bv, QT, KB, VT);
  k_fold<<<dim3(32, H_, B_), 128, 0, stream>>>(QT, MT, QP);
  k_attn<<<dim3(16, H_, B_), 256, 0, stream>>>(QP, KB, VT, TBL, out);
}
